// RGCN4_82154134438205
// MI455X (gfx1250) — matrix operand order repaired and checked by reading, not compiled, not run on hardware
//
#include <hip/hip_runtime.h>
#include <hip/hip_bf16.h>
#include <math.h>

// Problem constants (fixed by the reference)
constexpr int Nn   = 50000;
constexpr int Ee   = 800000;
constexpr int Rr   = 2;
constexpr int Ll   = 4;
constexpr int INd  = 64;
constexpr int HIDd = 128;
constexpr int Hh   = 8;
constexpr int Dd   = 16;
constexpr int OUTd = 32;
constexpr float EPSv = 1e-5f;

typedef __attribute__((ext_vector_type(16))) __bf16        v16bf;
typedef __attribute__((ext_vector_type(8)))  float         v8f;
typedef __attribute__((ext_vector_type(4)))  unsigned int  u32x4;

union Frag { v16bf v; u32x4 q[2]; };

__device__ __forceinline__ unsigned short f2bf(float f) {
  unsigned int u = __float_as_uint(f);
  u += 0x7FFFu + ((u >> 16) & 1u);   // round-to-nearest-even
  return (unsigned short)(u >> 16);
}

// ---------------------------------------------------------------------------
// bf16 WMMA GEMM: C[M,Nc] = A[M,K] @ B[K,Nc] (+ optional residual)
// A, B are bf16 (stored as u16), row-major. One wave computes one 16x16 tile.
// A frag: lane(0-15)=row M, hi selects K {0..7,16..23}/{8..15,24..31}
// B frag: lane(0-15)=column N, hi selects K {0..7,16..23}/{8..15,24..31}
// ---------------------------------------------------------------------------
__global__ __launch_bounds__(256)
void k_gemm(const unsigned short* __restrict__ A,
            const unsigned short* __restrict__ B,
            const float* __restrict__ resid,
            float* __restrict__ C,
            int K, int Nc, int numTiles) {
  int wid = blockIdx.x * (blockDim.x >> 5) + (threadIdx.x >> 5);
  if (wid >= numTiles) return;
  int tpr  = Nc >> 4;
  int tm   = wid / tpr, tn = wid % tpr;
  int lane = threadIdx.x & 31;
  int mr   = lane & 15, hi = lane >> 4;

  const unsigned short* arow  = A + (size_t)(tm * 16 + mr) * K;
  const unsigned short* bbase = B + tn * 16 + mr;

  v8f c = {};
  for (int k0 = 0; k0 < K; k0 += 32) {
    Frag a, b;
    const unsigned short* ap = arow + k0 + hi * 8;
    a.q[0] = *(const u32x4*)(ap);
    a.q[1] = *(const u32x4*)(ap + 16);
    const unsigned short* bp = bbase + (size_t)(k0 + hi * 8) * Nc;
    for (int s = 0; s < 16; s++)
      ((unsigned short*)&b)[s] = bp[(size_t)((s & 7) + (s >> 3) * 16) * Nc];
    c = __builtin_amdgcn_wmma_f32_16x16x32_bf16(false, a.v, false, b.v,
                                                (short)0, c, false, false);
  }
  int col   = tn * 16 + mr;
  int rbase = tm * 16 + hi * 8;
#pragma unroll
  for (int i = 0; i < 8; i++) {
    size_t idx = (size_t)(rbase + i) * Nc + col;
    float v = c[i];
    if (resid) v += resid[idx];
    C[idx] = v;
  }
}

// ---------------------------------------------------------------------------
// Elementwise helpers
// ---------------------------------------------------------------------------
__global__ void k_f2bf(const float* __restrict__ x, unsigned short* __restrict__ y, int n) {
  int i = blockIdx.x * blockDim.x + threadIdx.x;
  if (i < n) y[i] = f2bf(x[i]);
}

__global__ void k_fill(float* __restrict__ p, float v, int n) {
  int i = blockIdx.x * blockDim.x + threadIdx.x;
  if (i < n) p[i] = v;
}

// BatchNorm pass 1: per-column sum / sumsq with coalesced column mapping
__global__ void k_bn_stats(const float* __restrict__ x, float* __restrict__ stats,
                           int rows, int rowsPerBlock) {
  int col  = threadIdx.x & (HIDd - 1);
  int half = threadIdx.x >> 7;                 // 0..1 (256 threads)
  int r0   = blockIdx.x * rowsPerBlock + half;
  int rend = blockIdx.x * rowsPerBlock + rowsPerBlock;
  if (rend > rows) rend = rows;
  float s = 0.f, s2 = 0.f;
  for (int r = r0; r < rend; r += 2) {
    float v = x[(size_t)r * HIDd + col];
    s += v; s2 += v * v;
  }
  atomicAdd(&stats[col], s);
  atomicAdd(&stats[HIDd + col], s2);
}

// BatchNorm pass 2 + ReLU, in-place; also writes bf16 copy for next GEMM
__global__ void k_bn_apply(float* __restrict__ x, unsigned short* __restrict__ xb,
                           const float* __restrict__ stats,
                           const float* __restrict__ gamma,
                           const float* __restrict__ beta, int n) {
  int i = blockIdx.x * blockDim.x + threadIdx.x;
  if (i >= n) return;
  int c = i & (HIDd - 1);
  const float invN = 1.f / (float)Nn;
  float mean = stats[c] * invN;
  float var  = stats[HIDd + c] * invN - mean * mean;
  float v = (x[i] - mean) * rsqrtf(var + EPSv) * gamma[c] + beta[c];
  v = v > 0.f ? v : 0.f;
  x[i]  = v;
  xb[i] = f2bf(v);
}

// acc init with summed relation biases
__global__ void k_init_acc(float* __restrict__ acc, const float* __restrict__ b0,
                           const float* __restrict__ b1, int n) {
  int i = blockIdx.x * blockDim.x + threadIdx.x;
  if (i >= n) return;
  int c = i & (HIDd - 1);
  acc[i] = b0[c] + b1[c];
}

// leaky_relu(acc) + h1 -> h1 (residual), plus bf16 copy
__global__ void k_leaky_res(const float* __restrict__ acc, float* __restrict__ h1,
                            unsigned short* __restrict__ hb, int n) {
  int i = blockIdx.x * blockDim.x + threadIdx.x;
  if (i >= n) return;
  float a = acc[i];
  float v = a > 0.f ? a : 0.01f * a;
  float nh = v + h1[i];
  h1[i] = nh;
  hb[i] = f2bf(nh);
}

// ---------------------------------------------------------------------------
// GAT attention kernels
// ---------------------------------------------------------------------------
__device__ __forceinline__ float dot4(float4 a, float4 b) {
  return a.x * b.x + a.y * b.y + a.z * b.z + a.w * b.w;
}

// per (node, head): el = <feat, al>, er = <feat, ar>
__global__ void k_el_er(const float* __restrict__ feat,
                        const float* __restrict__ al, const float* __restrict__ ar,
                        float* __restrict__ el, float* __restrict__ er, int n) {
  int i = blockIdx.x * blockDim.x + threadIdx.x;
  if (i >= n) return;                           // n = N*H
  int h = i & (Hh - 1);
  const float4* f = (const float4*)(feat + (size_t)i * Dd);
  const float4* a = (const float4*)(al + h * Dd);
  const float4* b = (const float4*)(ar + h * Dd);
  float sl = 0.f, sr = 0.f;
#pragma unroll
  for (int j = 0; j < 4; j++) {
    float4 fv = f[j];
    sl += dot4(fv, a[j]);
    sr += dot4(fv, b[j]);
  }
  el[i] = sl;
  er[i] = sr;
}

// float atomic max via signed/unsigned int atomics (bit-order trick)
__device__ __forceinline__ void atomicMaxF(float* addr, float val) {
  if (val >= 0.f) atomicMax((int*)addr, __float_as_int(val));
  else            atomicMin((unsigned int*)addr, __float_as_uint(val));
}

__device__ __forceinline__ float lrelu02(float v) { return v > 0.f ? v : 0.2f * v; }

// per edge: segment max of attention logits over dst
__global__ void k_edge_max(const int* __restrict__ src, const int* __restrict__ dst,
                           const float* __restrict__ el, const float* __restrict__ er,
                           float* __restrict__ nmax, int ne) {
  int e = blockIdx.x * blockDim.x + threadIdx.x;
  if (e >= ne) return;
  int s = src[e], d = dst[e];
#pragma unroll
  for (int h = 0; h < Hh; h++) {
    float v = lrelu02(el[s * Hh + h] + er[d * Hh + h]);
    atomicMaxF(&nmax[d * Hh + h], v);
  }
}

// per edge: exp(e - max) accumulated into per-dst denominator
__global__ void k_edge_expsum(const int* __restrict__ src, const int* __restrict__ dst,
                              const float* __restrict__ el, const float* __restrict__ er,
                              const float* __restrict__ nmax, float* __restrict__ nsum,
                              int ne) {
  int e = blockIdx.x * blockDim.x + threadIdx.x;
  if (e >= ne) return;
  int s = src[e], d = dst[e];
#pragma unroll
  for (int h = 0; h < Hh; h++) {
    float v  = lrelu02(el[s * Hh + h] + er[d * Hh + h]);
    float ex = __expf(v - nmax[d * Hh + h]);
    atomicAdd(&nsum[d * Hh + h], ex);
  }
}

// per (edge, head): alpha * feat[src] scattered into acc[dst] (16 f32 atomics)
__global__ void k_scatter(const int* __restrict__ src, const int* __restrict__ dst,
                          const float* __restrict__ el, const float* __restrict__ er,
                          const float* __restrict__ nmax, const float* __restrict__ nsum,
                          const float* __restrict__ feat, float* __restrict__ acc, int n) {
  int i = blockIdx.x * blockDim.x + threadIdx.x;
  if (i >= n) return;                           // n = E*H
  int e = i >> 3, h = i & (Hh - 1);
  int s = src[e], d = dst[e];
  float v     = lrelu02(el[s * Hh + h] + er[d * Hh + h]);
  float alpha = __expf(v - nmax[d * Hh + h]) / nsum[d * Hh + h];
  const float4* f  = (const float4*)(feat + (size_t)s * HIDd + h * Dd);
  float*        op = acc + (size_t)d * HIDd + h * Dd;
#pragma unroll
  for (int j = 0; j < 4; j++) {
    float4 t = f[j];
    atomicAdd(op + 4 * j + 0, t.x * alpha);
    atomicAdd(op + 4 * j + 1, t.y * alpha);
    atomicAdd(op + 4 * j + 2, t.z * alpha);
    atomicAdd(op + 4 * j + 3, t.w * alpha);
  }
}

// ---------------------------------------------------------------------------
// Host driver
// ---------------------------------------------------------------------------
extern "C" void kernel_launch(void* const* d_in, const int* in_sizes, int n_in,
                              void* d_out, int out_size, void* d_ws, size_t ws_size,
                              hipStream_t stream) {
  const float* inputs      = (const float*)d_in[0];
  const int*   src         = (const int*)d_in[1];
  const int*   dstp        = (const int*)d_in[2];
  const float* embed_w1    = (const float*)d_in[3];
  const float* embed_gamma = (const float*)d_in[4];
  const float* embed_beta  = (const float*)d_in[5];
  const float* embed_w2    = (const float*)d_in[6];
  const float* gat_fc      = (const float*)d_in[7];
  const float* gat_al      = (const float*)d_in[8];
  const float* gat_ar      = (const float*)d_in[9];
  const float* gat_bias    = (const float*)d_in[10];
  const float* dec_w1      = (const float*)d_in[11];
  const float* dec_gamma   = (const float*)d_in[12];
  const float* dec_beta    = (const float*)d_in[13];
  const float* dec_w2      = (const float*)d_in[14];
  float* out = (float*)d_out;

  // workspace carve-out (256B aligned)
  char* p = (char*)d_ws;
  auto take = [&](size_t bytes) -> char* {
    char* r = p;
    p += (bytes + 255) & ~(size_t)255;
    return r;
  };
  float*          h1    = (float*)take((size_t)Nn * HIDd * 4);
  float*          tmp   = (float*)take((size_t)Nn * HIDd * 4); // also 'acc' (disjoint lifetime)
  float*          feat  = (float*)take((size_t)Nn * HIDd * 4);
  float*          el    = (float*)take((size_t)Nn * Hh * 4);
  float*          er    = (float*)take((size_t)Nn * Hh * 4);
  float*          nmax  = (float*)take((size_t)Nn * Hh * 4);
  float*          nsum  = (float*)take((size_t)Nn * Hh * 4);
  float*          stats = (float*)take(2 * HIDd * 4);
  unsigned short* hb    = (unsigned short*)take((size_t)Nn * HIDd * 2);
  unsigned short* w1b   = (unsigned short*)take((size_t)INd * HIDd * 2);
  unsigned short* w2b   = (unsigned short*)take((size_t)HIDd * HIDd * 2);
  unsigned short* gatb  = (unsigned short*)take((size_t)Ll * Rr * HIDd * HIDd * 2);
  unsigned short* d1b   = (unsigned short*)take((size_t)HIDd * HIDd * 2);
  unsigned short* d2b   = (unsigned short*)take((size_t)HIDd * OUTd * 2);
  float* acc = tmp;

  const int TPB = 256;
  auto blocks = [](int n) { return (n + 255) / 256; };
  auto gemm = [&](const unsigned short* A, const unsigned short* B,
                  const float* resid, float* C, int K, int Nc) {
    int tiles = (Nn / 16) * (Nc / 16);
    k_gemm<<<(tiles + 7) / 8, 256, 0, stream>>>(A, B, resid, C, K, Nc, tiles);
  };

  // ---- weight conversion to bf16 (row-major, matches WMMA B fragment loads)
  k_f2bf<<<blocks(INd * HIDd), TPB, 0, stream>>>(embed_w1, w1b, INd * HIDd);
  k_f2bf<<<blocks(HIDd * HIDd), TPB, 0, stream>>>(embed_w2, w2b, HIDd * HIDd);
  k_f2bf<<<blocks(Ll * Rr * HIDd * HIDd), TPB, 0, stream>>>(gat_fc, gatb, Ll * Rr * HIDd * HIDd);
  k_f2bf<<<blocks(HIDd * HIDd), TPB, 0, stream>>>(dec_w1, d1b, HIDd * HIDd);
  k_f2bf<<<blocks(HIDd * OUTd), TPB, 0, stream>>>(dec_w2, d2b, HIDd * OUTd);

  // ---- embed MLP: h = relu(bn(x@w1)); h1 = h@w2 + h
  k_f2bf<<<blocks(Nn * INd), TPB, 0, stream>>>(inputs, hb, Nn * INd);
  gemm(hb, w1b, nullptr, tmp, INd, HIDd);
  k_fill<<<1, TPB, 0, stream>>>(stats, 0.f, 2 * HIDd);
  k_bn_stats<<<(Nn + 511) / 512, TPB, 0, stream>>>(tmp, stats, Nn, 512);
  k_bn_apply<<<blocks(Nn * HIDd), TPB, 0, stream>>>(tmp, hb, stats, embed_gamma, embed_beta, Nn * HIDd);
  gemm(hb, w2b, /*resid=*/tmp, h1, HIDd, HIDd);
  k_f2bf<<<blocks(Nn * HIDd), TPB, 0, stream>>>(h1, hb, Nn * HIDd);

  // ---- GAT layers
  for (int l = 0; l < Ll; l++) {
    k_init_acc<<<blocks(Nn * HIDd), TPB, 0, stream>>>(
        acc, gat_bias + (size_t)(l * Rr + 0) * HIDd,
             gat_bias + (size_t)(l * Rr + 1) * HIDd, Nn * HIDd);
    for (int r = 0; r < Rr; r++) {
      const int*   se = src  + (size_t)r * Ee;
      const int*   de = dstp + (size_t)r * Ee;
      const float* al = gat_al + (size_t)(l * Rr + r) * Hh * Dd;
      const float* ar = gat_ar + (size_t)(l * Rr + r) * Hh * Dd;
      gemm(hb, gatb + (size_t)(l * Rr + r) * HIDd * HIDd, nullptr, feat, HIDd, HIDd);
      k_el_er<<<blocks(Nn * Hh), TPB, 0, stream>>>(feat, al, ar, el, er, Nn * Hh);
      k_fill<<<blocks(Nn * Hh), TPB, 0, stream>>>(nmax, -__builtin_inff(), Nn * Hh);
      k_fill<<<blocks(Nn * Hh), TPB, 0, stream>>>(nsum, 0.f, Nn * Hh);
      k_edge_max<<<blocks(Ee), TPB, 0, stream>>>(se, de, el, er, nmax, Ee);
      k_edge_expsum<<<blocks(Ee), TPB, 0, stream>>>(se, de, el, er, nmax, nsum, Ee);
      k_scatter<<<blocks(Ee * Hh), TPB, 0, stream>>>(se, de, el, er, nmax, nsum, feat, acc, Ee * Hh);
    }
    k_leaky_res<<<blocks(Nn * HIDd), TPB, 0, stream>>>(acc, h1, hb, Nn * HIDd);
  }

  // ---- decision MLP: out = relu(bn(h1@w1)) @ w2
  gemm(hb, d1b, nullptr, tmp, HIDd, HIDd);
  k_fill<<<1, TPB, 0, stream>>>(stats, 0.f, 2 * HIDd);
  k_bn_stats<<<(Nn + 511) / 512, TPB, 0, stream>>>(tmp, stats, Nn, 512);
  k_bn_apply<<<blocks(Nn * HIDd), TPB, 0, stream>>>(tmp, hb, stats, dec_gamma, dec_beta, Nn * HIDd);
  gemm(hb, d2b, nullptr, out, HIDd, OUTd);
}
